// ECGMambaClassifier_90460601189075
// MI455X (gfx1250) — compile-verified, matrix operand order checked by reading, not run on hardware
//
#include <hip/hip_runtime.h>

typedef __attribute__((ext_vector_type(16))) __bf16 v16bf;
typedef __attribute__((ext_vector_type(8)))  float  v8f;

constexpr int CB     = 16;                     // batch
constexpr int CL     = 1000;                   // sequence length
constexpr int CT     = CB * CL;                // tokens = 16000
constexpr int CDM    = 256;
constexpr int CDIN   = 512;
constexpr int CNS    = 64;
constexpr int CNH    = 8;
constexpr int CPH    = 64;
constexpr int CCONVD = CDIN + 2 * CNS;         // 640
constexpr int CKW    = 4;
constexpr int CNL    = 6;
constexpr int CHID   = 1024;
constexpr int CDPROJ = 2 * CDIN + 2 * CNS + CNH; // 1160

__device__ __forceinline__ float siluf(float x) { return x / (1.f + expf(-x)); }

__device__ __forceinline__ float4 ld4(const float* p) {
  return *reinterpret_cast<const float4*>(p);
}

// Convert 8 consecutive f32 (two float4) into fragment elements [base, base+8)
// using the native f32->bf16 conversion (RNE).
__device__ __forceinline__ void cvt8(v16bf& d, int base, float4 x, float4 y) {
  d[base + 0] = (__bf16)x.x; d[base + 1] = (__bf16)x.y;
  d[base + 2] = (__bf16)x.z; d[base + 3] = (__bf16)x.w;
  d[base + 4] = (__bf16)y.x; d[base + 5] = (__bf16)y.y;
  d[base + 6] = (__bf16)y.z; d[base + 7] = (__bf16)y.w;
}

// ---------------------------------------------------------------------------
// Generic GEMM: C[:, col_off + n] (beta?+=:=) sum_k A[m,k] * W[n,k] (+ bias[n])
// bf16 WMMA, f32 accumulate. Block tile = 64 rows x 128 cols, blockDim (32,8).
// Wave (wr,wc): rows wr*16..+16, cols wc*64..+64 -> 1 A frag, 4 B frags,
// 4 WMMAs per K-step. M (=CT) multiple of 64; K multiple of 32; N guarded.
// flipL != 0: output row r remapped to b*flipL + (flipL-1-l) within batch.
// ---------------------------------------------------------------------------
__global__ __launch_bounds__(256) void k_gemm(
    const float* __restrict__ A, int lda,
    const float* __restrict__ W, int N, int Kd,
    const float* __restrict__ bias,
    float* __restrict__ C, int ldc, int col_off,
    int beta, int flipL)
{
  const int lane = threadIdx.x;
  const int wave = threadIdx.y;
  const int wr = wave >> 1, wc = wave & 1;
  const int row0 = blockIdx.y * 64 + wr * 16;
  const int colBase = blockIdx.x * 128 + wc * 64;
  const int m = lane & 15, half = lane >> 4;

  const float* arow = A + (size_t)(row0 + m) * lda;
  const float* wrow[4];
  bool ok[4];
#pragma unroll
  for (int s = 0; s < 4; ++s) {
    int n = colBase + s * 16 + m;
    ok[s] = n < N;
    wrow[s] = W + (size_t)(ok[s] ? n : 0) * Kd;
  }

  v8f acc[4];
#pragma unroll
  for (int s = 0; s < 4; ++s) acc[s] = v8f{};

  for (int k0 = 0; k0 < Kd; k0 += 32) {
    const int koff = k0 + half * 8;
    v16bf af{};
    cvt8(af, 0, ld4(arow + koff),      ld4(arow + koff + 4));
    cvt8(af, 8, ld4(arow + koff + 16), ld4(arow + koff + 20));
    __builtin_prefetch(arow + koff + 32, 0, 1);
    v16bf bf[4];
#pragma unroll
    for (int s = 0; s < 4; ++s) {
      bf[s] = v16bf{};
      cvt8(bf[s], 0, ld4(wrow[s] + koff),      ld4(wrow[s] + koff + 4));
      cvt8(bf[s], 8, ld4(wrow[s] + koff + 16), ld4(wrow[s] + koff + 20));
    }
#pragma unroll
    for (int s = 0; s < 4; ++s) {
      acc[s] = __builtin_amdgcn_wmma_f32_16x16x32_bf16(false, af, false, bf[s],
                                                       (short)0, acc[s], false, false);
    }
  }

  // D layout: VGPR v holds row (row0 + v + 8*half), col = colBase + s*16 + (lane&15)
#pragma unroll
  for (int v = 0; v < 8; ++v) {
    int r = row0 + v + 8 * half;
    int orow = r;
    if (flipL) { int bb = r / flipL, ll = r % flipL; orow = bb * flipL + (flipL - 1 - ll); }
    float* crow = C + (size_t)orow * ldc + col_off;
#pragma unroll
    for (int s = 0; s < 4; ++s) {
      int cn = colBase + s * 16 + m;
      if (cn < N) {
        float val = acc[s][v] + (bias ? bias[cn] : 0.f);
        crow[cn] = beta ? (crow[cn] + val) : val;
      }
    }
  }
}

// ---------------------------------------------------------------------------
// Input projection: H[b,l,dm] = sum_c x[b,c,l] * W[dm,c] + bias[dm]
// ---------------------------------------------------------------------------
__global__ __launch_bounds__(256) void k_inproj(
    const float* __restrict__ x, const float* __restrict__ W,
    const float* __restrict__ bias, float* __restrict__ H)
{
  size_t idx = (size_t)blockIdx.x * 256 + threadIdx.x;
  if (idx >= (size_t)CT * CDM) return;
  int dm = (int)(idx % CDM);
  size_t row = idx / CDM;
  int b = (int)(row / CL), l = (int)(row % CL);
  float acc = bias[dm];
#pragma unroll
  for (int c = 0; c < 12; ++c)
    acc += x[((size_t)b * 12 + c) * CL + l] * W[dm * 12 + c];
  H[idx] = acc;
}

// RMS norm over DM=256, one token per block, one element per thread
__global__ __launch_bounds__(256) void k_rms(
    const float* __restrict__ in, const float* __restrict__ w, float* __restrict__ out)
{
  __shared__ float red[256];
  int row = blockIdx.x, t = threadIdx.x;
  float v = in[(size_t)row * CDM + t];
  red[t] = v * v; __syncthreads();
  for (int o = 128; o; o >>= 1) { if (t < o) red[t] += red[t + o]; __syncthreads(); }
  float inv = rsqrtf(red[0] / (float)CDM + 1e-5f);
  out[(size_t)row * CDM + t] = v * inv * w[t];
}

// Flip the sequence dimension within each batch (DM columns)
__global__ __launch_bounds__(256) void k_flip(
    const float* __restrict__ in, float* __restrict__ out)
{
  size_t idx = (size_t)blockIdx.x * 256 + threadIdx.x;
  if (idx >= (size_t)CT * CDM) return;
  int dm = (int)(idx % CDM);
  size_t row = idx / CDM;
  int b = (int)(row / CL), l = (int)(row % CL);
  size_t src = (size_t)b * CL + (CL - 1 - l);
  out[idx] = in[src * CDM + dm];
}

// Causal depthwise conv (K=4) + SiLU over the xBC slice of zxbcdt
__global__ __launch_bounds__(256) void k_conv_silu(
    const float* __restrict__ Z, const float* __restrict__ cw,
    const float* __restrict__ cb, float* __restrict__ xBC)
{
  size_t idx = (size_t)blockIdx.x * 256 + threadIdx.x;
  if (idx >= (size_t)CT * CCONVD) return;
  int c = (int)(idx % CCONVD);
  size_t row = idx / CCONVD;
  int b = (int)(row / CL), l = (int)(row % CL);
  float acc = cb[c];
#pragma unroll
  for (int k = 0; k < CKW; ++k) {
    int ls = l + k - (CKW - 1);
    if (ls >= 0)
      acc += cw[c * CKW + k] * Z[((size_t)b * CL + ls) * CDPROJ + CDIN + c];
  }
  xBC[idx] = siluf(acc);
}

// dt = softplus(z_dt + dtb)
__global__ __launch_bounds__(256) void k_dt(
    const float* __restrict__ Z, const float* __restrict__ dtb, float* __restrict__ dt)
{
  size_t idx = (size_t)blockIdx.x * 256 + threadIdx.x;
  if (idx >= (size_t)CT * CNH) return;
  int hh = (int)(idx % CNH);
  size_t row = idx / CNH;
  float x = Z[row * CDPROJ + CDIN + CCONVD + hh] + dtb[hh];
  dt[idx] = (x > 20.f) ? x : log1pf(expf(x));
}

// Sequential SSD scan. One block per (batch, head). 256 threads:
// thread t owns state h[p][nq..nq+15], p = t>>2, nq = (t&3)*16.
__global__ __launch_bounds__(256) void k_scan(
    const float* __restrict__ xBC, const float* __restrict__ dt,
    const float* __restrict__ Alog, const float* __restrict__ Dp,
    float* __restrict__ ys)
{
  __shared__ float sX[CPH], sB[CNS], sC[CNS];
  const int bh = blockIdx.x;
  const int b = bh / CNH, hh = bh % CNH;
  const int t = threadIdx.x;
  const int p = t >> 2, q = t & 3, nq = q * 16;
  const float Ah = -expf(Alog[hh]);
  const float Dh = Dp[hh];
  float st[16];
#pragma unroll
  for (int j = 0; j < 16; ++j) st[j] = 0.f;

  for (int l = 0; l < CL; ++l) {
    const size_t row = (size_t)b * CL + l;
    const float* xr = xBC + row * CCONVD;
    if (t < 64)        sX[t]       = xr[hh * CPH + t];
    else if (t < 128)  sB[t - 64]  = xr[CDIN + (t - 64)];
    else if (t < 192)  sC[t - 128] = xr[CDIN + CNS + (t - 128)];
    __syncthreads();
    const float dtv = dt[row * CNH + hh];
    const float dA  = expf(dtv * Ah);
    const float xp  = sX[p];
    const float dBx = dtv * xp;
    float acc = 0.f;
#pragma unroll
    for (int j = 0; j < 16; ++j) {
      st[j] = st[j] * dA + dBx * sB[nq + j];
      acc += st[j] * sC[nq + j];
    }
    acc += __shfl_xor(acc, 1, 32);
    acc += __shfl_xor(acc, 2, 32);
    if (q == 0) ys[row * CDIN + hh * CPH + p] = acc + Dh * xp;
    __syncthreads();
  }
}

// y = rms(y * silu(z)) * nw, over DIN=512 (in-place on ys), 2 elems/thread
__global__ __launch_bounds__(256) void k_gate_rms(
    float* __restrict__ ys, const float* __restrict__ Z, const float* __restrict__ nw)
{
  __shared__ float red[256];
  int row = blockIdx.x, t = threadIdx.x;
  size_t zb = (size_t)row * CDPROJ;
  size_t yb = (size_t)row * CDIN;
  float z0 = Z[zb + t], z1 = Z[zb + 256 + t];
  float v0 = ys[yb + t]       * siluf(z0);
  float v1 = ys[yb + 256 + t] * siluf(z1);
  red[t] = v0 * v0 + v1 * v1; __syncthreads();
  for (int o = 128; o; o >>= 1) { if (t < o) red[t] += red[t + o]; __syncthreads(); }
  float inv = rsqrtf(red[0] / (float)CDIN + 1e-5f);
  ys[yb + t]       = v0 * inv * nw[t];
  ys[yb + 256 + t] = v1 * inv * nw[256 + t];
}

// GLU: g = silu(u) * v   (u = uv[:,:HID], v = uv[:,HID:]), uv stride 2*HID
__global__ __launch_bounds__(256) void k_glu(
    const float* __restrict__ UV, float* __restrict__ G)
{
  size_t idx = (size_t)blockIdx.x * 256 + threadIdx.x;
  if (idx >= (size_t)CT * CHID) return;
  int i = (int)(idx % CHID);
  size_t row = idx / CHID;
  float u = UV[row * (2 * CHID) + i];
  float v = UV[row * (2 * CHID) + CHID + i];
  G[idx] = siluf(u) * v;
}

// Softmax attention pool + head. One block per batch.
__global__ __launch_bounds__(256) void k_pool_head(
    const float* __restrict__ xn, const float* __restrict__ pw,
    const float* __restrict__ pb, const float* __restrict__ hw,
    const float* __restrict__ hb, float* __restrict__ out)
{
  __shared__ float s[CL];
  __shared__ float red[256];
  int b = blockIdx.x, t = threadIdx.x;
  for (int l = t; l < CL; l += 256) {
    const float* r = xn + (size_t)(b * CL + l) * CDM;
    float acc = 0.f;
    for (int d = 0; d < CDM; ++d) acc += r[d] * pw[d];
    s[l] = acc + pb[0];
  }
  __syncthreads();
  float mx = -1e30f;
  for (int l = t; l < CL; l += 256) mx = fmaxf(mx, s[l]);
  red[t] = mx; __syncthreads();
  for (int o = 128; o; o >>= 1) { if (t < o) red[t] = fmaxf(red[t], red[t + o]); __syncthreads(); }
  mx = red[0]; __syncthreads();
  float sm = 0.f;
  for (int l = t; l < CL; l += 256) { float e = expf(s[l] - mx); s[l] = e; sm += e; }
  red[t] = sm; __syncthreads();
  for (int o = 128; o; o >>= 1) { if (t < o) red[t] += red[t + o]; __syncthreads(); }
  sm = red[0]; __syncthreads();
  float pv = 0.f;                            // pooled value for feature dim t
  for (int l = 0; l < CL; ++l) pv += xn[(size_t)(b * CL + l) * CDM + t] * s[l];
  pv /= sm;
  red[t] = pv * hw[t]; __syncthreads();
  for (int o = 128; o; o >>= 1) { if (t < o) red[t] += red[t + o]; __syncthreads(); }
  if (t == 0) out[b] = red[0] + hb[0];
}

// ---------------------------------------------------------------------------
static void launch_gemm(hipStream_t st, const float* A, int lda,
                        const float* W, int N, int Kd, const float* bias,
                        float* C, int ldc, int col_off, int beta, int flipL)
{
  dim3 blk(32, 8);
  dim3 grd((N + 127) / 128, CT / 64);
  k_gemm<<<grd, blk, 0, st>>>(A, lda, W, N, Kd, bias, C, ldc, col_off, beta, flipL);
}

static void mamba_dir(hipStream_t st, const float* xin,
                      const float* Win, const float* cw, const float* cb,
                      const float* dtb, const float* Alog, const float* Dp,
                      const float* nw, const float* Wout,
                      float* BIG, float* XBC, float* DT, float* YS,
                      float* YCAT, int col_off, int flipL)
{
  // zxbcdt = xin @ Win^T
  launch_gemm(st, xin, CDM, Win, CDPROJ, CDM, nullptr, BIG, CDPROJ, 0, 0, 0);
  {
    size_t n = (size_t)CT * CCONVD;
    k_conv_silu<<<(unsigned)((n + 255) / 256), 256, 0, st>>>(BIG, cw, cb, XBC);
  }
  {
    size_t n = (size_t)CT * CNH;
    k_dt<<<(unsigned)((n + 255) / 256), 256, 0, st>>>(BIG, dtb, DT);
  }
  k_scan<<<CB * CNH, 256, 0, st>>>(XBC, DT, Alog, Dp, YS);
  k_gate_rms<<<CT, 256, 0, st>>>(YS, BIG, nw);
  // out proj into YCAT half (flip rows for backward direction)
  launch_gemm(st, YS, CDIN, Wout, CDM, CDIN, nullptr, YCAT, 2 * CDM, col_off, 0, flipL);
}

extern "C" void kernel_launch(void* const* d_in, const int* in_sizes, int n_in,
                              void* d_out, int out_size, void* d_ws, size_t ws_size,
                              hipStream_t stream) {
  (void)in_sizes; (void)n_in; (void)out_size; (void)ws_size;
  const float* x         = (const float*)d_in[0];
  const float* in_proj_w = (const float*)d_in[1];
  const float* in_proj_b = (const float*)d_in[2];
  const float* norm1_w   = (const float*)d_in[3];
  const float* norm2_w   = (const float*)d_in[4];
  const float* m_in_w_f   = (const float*)d_in[5];
  const float* m_conv_w_f = (const float*)d_in[6];
  const float* m_conv_b_f = (const float*)d_in[7];
  const float* m_dtb_f    = (const float*)d_in[8];
  const float* m_Alog_f   = (const float*)d_in[9];
  const float* m_D_f      = (const float*)d_in[10];
  const float* m_norm_w_f = (const float*)d_in[11];
  const float* m_out_w_f  = (const float*)d_in[12];
  const float* m_in_w_b   = (const float*)d_in[13];
  const float* m_conv_w_b = (const float*)d_in[14];
  const float* m_conv_b_b = (const float*)d_in[15];
  const float* m_dtb_b    = (const float*)d_in[16];
  const float* m_Alog_b   = (const float*)d_in[17];
  const float* m_D_b      = (const float*)d_in[18];
  const float* m_norm_w_b = (const float*)d_in[19];
  const float* m_out_w_b  = (const float*)d_in[20];
  const float* bi_w   = (const float*)d_in[21];
  const float* bi_b   = (const float*)d_in[22];
  const float* fc1_w  = (const float*)d_in[23];
  const float* fc1_b  = (const float*)d_in[24];
  const float* fc2_w  = (const float*)d_in[25];
  const float* fc2_b  = (const float*)d_in[26];
  const float* norm_out_w = (const float*)d_in[27];
  const float* pool_w = (const float*)d_in[28];
  const float* pool_b = (const float*)d_in[29];
  const float* head_w = (const float*)d_in[30];
  const float* head_b = (const float*)d_in[31];

  // Workspace arena (f32)
  float* ws = (float*)d_ws;
  size_t off = 0;
  float* H    = ws + off; off += (size_t)CT * CDM;
  float* XN   = ws + off; off += (size_t)CT * CDM;
  float* XN2  = ws + off; off += (size_t)CT * CDM;
  float* BIG  = ws + off; off += (size_t)CT * (2 * CHID); // zxbcdt / uv
  float* XBC  = ws + off; off += (size_t)CT * CCONVD;
  float* DT   = ws + off; off += (size_t)CT * CNH;
  float* YS   = ws + off; off += (size_t)CT * CDIN;
  float* YCAT = ws + off; off += (size_t)CT * (2 * CDM);
  float* GLU  = ws + off; off += (size_t)CT * CHID;

  // h = x^T @ in_proj_w^T + b
  {
    size_t n = (size_t)CT * CDM;
    k_inproj<<<(unsigned)((n + 255) / 256), 256, 0, stream>>>(x, in_proj_w, in_proj_b, H);
  }

  for (int layer = 0; layer < CNL; ++layer) {
    k_rms<<<CT, 256, 0, stream>>>(H, norm1_w + (size_t)layer * CDM, XN);

    // forward direction -> YCAT[:, 0:256]
    mamba_dir(stream, XN,
              m_in_w_f   + (size_t)layer * CDPROJ * CDM,
              m_conv_w_f + (size_t)layer * CCONVD * CKW,
              m_conv_b_f + (size_t)layer * CCONVD,
              m_dtb_f    + (size_t)layer * CNH,
              m_Alog_f   + (size_t)layer * CNH,
              m_D_f      + (size_t)layer * CNH,
              m_norm_w_f + (size_t)layer * CDIN,
              m_out_w_f  + (size_t)layer * CDM * CDIN,
              BIG, XBC, DT, YS, YCAT, 0, 0);

    // backward direction: flip, run, flip back at store -> YCAT[:, 256:512]
    {
      size_t n = (size_t)CT * CDM;
      k_flip<<<(unsigned)((n + 255) / 256), 256, 0, stream>>>(XN, XN2);
    }
    mamba_dir(stream, XN2,
              m_in_w_b   + (size_t)layer * CDPROJ * CDM,
              m_conv_w_b + (size_t)layer * CCONVD * CKW,
              m_conv_b_b + (size_t)layer * CCONVD,
              m_dtb_b    + (size_t)layer * CNH,
              m_Alog_b   + (size_t)layer * CNH,
              m_D_b      + (size_t)layer * CNH,
              m_norm_w_b + (size_t)layer * CDIN,
              m_out_w_b  + (size_t)layer * CDM * CDIN,
              BIG, XBC, DT, YS, YCAT, CDM, CL);

    // h += [y_f, y_b] @ bi_w^T + bi_b
    launch_gemm(stream, YCAT, 2 * CDM,
                bi_w + (size_t)layer * CDM * 2 * CDM, CDM, 2 * CDM,
                bi_b + (size_t)layer * CDM, H, CDM, 0, 1, 0);

    // MLP
    k_rms<<<CT, 256, 0, stream>>>(H, norm2_w + (size_t)layer * CDM, XN);
    launch_gemm(stream, XN, CDM,
                fc1_w + (size_t)layer * 2 * CHID * CDM, 2 * CHID, CDM,
                fc1_b + (size_t)layer * 2 * CHID, BIG, 2 * CHID, 0, 0, 0);
    {
      size_t n = (size_t)CT * CHID;
      k_glu<<<(unsigned)((n + 255) / 256), 256, 0, stream>>>(BIG, GLU);
    }
    launch_gemm(stream, GLU, CHID,
                fc2_w + (size_t)layer * CDM * CHID, CDM, CHID,
                fc2_b + (size_t)layer * CDM, H, CDM, 0, 1, 0);
  }

  // final norm + softmax pool + head
  k_rms<<<CT, 256, 0, stream>>>(H, norm_out_w, XN);
  k_pool_head<<<CB, 256, 0, stream>>>(XN, pool_w, pool_b, head_w, head_b, (float*)d_out);
}
